// Perceiver_48129403518991
// MI455X (gfx1250) — matrix operand order repaired and checked by reading, not compiled, not run on hardware
//
#include <hip/hip_runtime.h>
#include <hip/hip_bf16.h>

// ---------------- model constants ----------------
#define DEPTH 24
#define DLAT  80
#define HEADS 8
#define DH    64
#define INNER 512      // HEADS*DH
#define FFD   160
#define BB    64
#define NN    128
#define MM    (BB*NN)  // 8192 tokens
#define MHD   20
#define KPAD  96       // 80 padded to multiple of 32

typedef __attribute__((ext_vector_type(16))) _Float16 v16h;
typedef __attribute__((ext_vector_type(8)))  _Float16 v8h;
typedef __attribute__((ext_vector_type(8)))  float    v8f;

static __device__ __forceinline__ v16h mk16(v8h lo, v8h hi) {
  v16h r;
#pragma unroll
  for (int i = 0; i < 8; ++i) { r[i] = lo[i]; r[i + 8] = hi[i]; }
  return r;
}

// ---------------- graph structural bias ----------------
// bias[b,i,j] = sum_hop mean_f(edge_emb[edge_input]) * edge_dis_w[hop] / sp + spatial_emb[sp_pos]
__global__ void bias_kernel(const int* __restrict__ sp_pos, const int* __restrict__ edge_in,
                            const float* __restrict__ edge_emb, const float* __restrict__ edge_dis_w,
                            const float* __restrict__ spatial_emb, float* __restrict__ bias) {
  int idx = blockIdx.x * 256 + threadIdx.x;
  if (idx >= BB * NN * NN) return;
  int spr = sp_pos[idx];
  float spb = spatial_emb[spr];
  int sp = (spr == 0) ? 1 : spr;
  if (sp > 1) sp -= 1;
  if (sp > MHD) sp = MHD;
  const int* ep = edge_in + (size_t)idx * (MHD * 3);
  float e = 0.f;
#pragma unroll 4
  for (int hop = 0; hop < MHD; ++hop) {
    float s = edge_emb[ep[hop * 3 + 0]] + edge_emb[ep[hop * 3 + 1]] + edge_emb[ep[hop * 3 + 2]];
    e += (s * (1.f / 3.f)) * edge_dis_w[hop];
  }
  bias[idx] = e / (float)sp + spb;
}

// ---------------- node embeddings ----------------
__global__ void embed_kernel(const int* __restrict__ xn, const int* __restrict__ ind,
                             const int* __restrict__ outd, const float* __restrict__ aemb,
                             const float* __restrict__ iemb, const float* __restrict__ oemb,
                             float* __restrict__ x) {
  int idx = blockIdx.x * 256 + threadIdx.x;
  if (idx >= MM * DLAT) return;
  int d = idx % DLAT; int bn = idx / DLAT;
  x[idx] = aemb[xn[bn] * DLAT + d] + iemb[ind[bn] * DLAT + d] + oemb[outd[bn] * DLAT + d];
}

// ---------------- weight convert f32 [D,K,N] -> f16 transposed [D,N,Kpad], zero pad ----------------
__global__ void wconv_kernel(const float* __restrict__ W, _Float16* __restrict__ Wt,
                             int K, int Kpad, int NC, int total) {
  int idx = blockIdx.x * 256 + threadIdx.x;
  if (idx >= total) return;
  int kk = idx % Kpad; int r = idx / Kpad;
  int n = r % NC; int l = r / NC;
  float v = (kk < K) ? W[((size_t)l * K + kk) * NC + n] : 0.f;
  Wt[idx] = (_Float16)v;
}

// ---------------- layernorm(row of 80) -> f16 row of 96 (zero-padded) ----------------
__global__ void ln_rows_f16(const float* __restrict__ x, const float* __restrict__ g,
                            const float* __restrict__ be, _Float16* __restrict__ out) {
  __shared__ float red[128];
  int row = blockIdx.x, t = threadIdx.x;
  float v = (t < DLAT) ? x[(size_t)row * DLAT + t] : 0.f;
  red[t] = v; __syncthreads();
  for (int s = 64; s > 0; s >>= 1) { if (t < s) red[t] += red[t + s]; __syncthreads(); }
  float mean = red[0] * (1.f / DLAT); __syncthreads();
  float d = (t < DLAT) ? v - mean : 0.f;
  red[t] = d * d; __syncthreads();
  for (int s = 64; s > 0; s >>= 1) { if (t < s) red[t] += red[t + s]; __syncthreads(); }
  float rstd = rsqrtf(red[0] * (1.f / DLAT) + 1e-5f);
  if (t < KPAD)
    out[(size_t)row * KPAD + t] = (t < DLAT) ? (_Float16)(d * rstd * g[t] + be[t]) : (_Float16)0.f;
}

// ---------------- WMMA GEMM, M-blocked 4x per wave ----------------
// C[M,NC] = A[M,K] @ Wt^T  (Wt is [NC][K] f16). Each wave: one 16-wide N tile,
// four 16-tall M subtiles sharing the B (weight) fragment -> 2.5 loads per WMMA.
// Epilogue: +bias, optional exact GELU, f16 store (outh) and/or f32 residual add (resid).
__global__ void gemm_f16_wmma(const _Float16* __restrict__ A, int lda,
                              const _Float16* __restrict__ Wt, int K, int NC,
                              const float* __restrict__ bias,
                              float* resid, _Float16* __restrict__ outh,
                              int ldo, int gelu) {
  int wave = threadIdx.x >> 5, lane = threadIdx.x & 31;
  int ml = lane & 15, h2 = lane >> 4;
  int m0 = blockIdx.x * 64;
  int n0 = (blockIdx.y * 4 + wave) * 16;
  if (n0 >= NC) return;                 // wave-uniform: EXEC stays all-ones for WMMA
  const _Float16* brow  = Wt + (size_t)(n0 + ml) * K;
  const _Float16* arow0 = A + (size_t)(m0 + ml) * lda;
  const size_t astep = (size_t)16 * lda;
  v8f c0 = {}, c1 = {}, c2 = {}, c3 = {};
  for (int k0 = 0; k0 < K; k0 += 32) {
    v16h bf = mk16(*(const v8h*)(brow + k0 + 8 * h2),
                   *(const v8h*)(brow + k0 + 16 + 8 * h2));
    const _Float16* ap = arow0 + k0;
    v16h a0 = mk16(*(const v8h*)(ap + 8 * h2), *(const v8h*)(ap + 16 + 8 * h2)); ap += astep;
    v16h a1 = mk16(*(const v8h*)(ap + 8 * h2), *(const v8h*)(ap + 16 + 8 * h2)); ap += astep;
    v16h a2 = mk16(*(const v8h*)(ap + 8 * h2), *(const v8h*)(ap + 16 + 8 * h2)); ap += astep;
    v16h a3 = mk16(*(const v8h*)(ap + 8 * h2), *(const v8h*)(ap + 16 + 8 * h2));
    c0 = __builtin_amdgcn_wmma_f32_16x16x32_f16(false, a0, false, bf, (short)0, c0, false, false);
    c1 = __builtin_amdgcn_wmma_f32_16x16x32_f16(false, a1, false, bf, (short)0, c1, false, false);
    c2 = __builtin_amdgcn_wmma_f32_16x16x32_f16(false, a2, false, bf, (short)0, c2, false, false);
    c3 = __builtin_amdgcn_wmma_f32_16x16x32_f16(false, a3, false, bf, (short)0, c3, false, false);
  }
  int n = n0 + ml;
  float bv = bias ? bias[n] : 0.f;
  v8f acc[4] = {c0, c1, c2, c3};
#pragma unroll
  for (int s = 0; s < 4; ++s) {
#pragma unroll
    for (int r = 0; r < 8; ++r) {
      int m = m0 + s * 16 + r + 8 * h2;
      float vv = acc[s][r] + bv;
      if (gelu) vv = 0.5f * vv * (1.f + erff(vv * 0.70710678118f));  // exact gelu
      size_t o = (size_t)m * ldo + n;
      if (resid) resid[o] += vv;        // x = proj + bias + x
      if (outh)  outh[o] = (_Float16)vv;
    }
  }
}

// ---------------- fused attention: one block per (b, h, 16-query tile) ----------------
// S = Q K^T * 0.125 + bias ; P = softmax(S) ; O = P V
__global__ void attn_kernel(const _Float16* __restrict__ q,    // [MM, INNER]
                            const _Float16* __restrict__ kv,   // [MM, 2*INNER] (k | v)
                            const float* __restrict__ bias,    // [BB, NN, NN]
                            _Float16* __restrict__ o) {        // [MM, INNER]
  __shared__ _Float16 Vt[DH][NN];      // V transposed: Vt[d][j]
  __shared__ float    Ssm[16][NN];
  __shared__ _Float16 Pm[16][NN];
  __shared__ float    rmax[16], rsum[16];
  int t = threadIdx.x;
  int wave = t >> 5, lane = t & 31;
  int ml = lane & 15, h2 = lane >> 4;
  int it = blockIdx.x & 7;
  int bh = blockIdx.x >> 3;
  int h = bh & 7, b = bh >> 3;
  size_t rowb = (size_t)b * NN;
  int hc = h * DH;
  // stage V^T into LDS (coalesced global read along d)
  for (int idx = t; idx < NN * DH; idx += 256) {
    int j = idx >> 6, d = idx & 63;
    Vt[d][j] = kv[(rowb + j) * (2 * INNER) + INNER + hc + d];
  }
  // S tiles: wave w owns key tile j0 = 16w
  {
    int j0 = wave * 16;
    const _Float16* qrow = q  + (rowb + it * 16 + ml) * INNER + hc;
    const _Float16* krow = kv + (rowb + j0 + ml) * (2 * INNER) + hc;
    v8f c = {};
#pragma unroll
    for (int dk = 0; dk < DH; dk += 32) {
      v8h a0 = *(const v8h*)(qrow + dk + 8 * h2);
      v8h a1 = *(const v8h*)(qrow + dk + 16 + 8 * h2);
      v8h b0 = *(const v8h*)(krow + dk + 8 * h2);
      v8h b1 = *(const v8h*)(krow + dk + 16 + 8 * h2);
      c = __builtin_amdgcn_wmma_f32_16x16x32_f16(false, mk16(a0, a1), false, mk16(b0, b1),
                                                 (short)0, c, false, false);
    }
    const float* bp = bias + (rowb + it * 16) * NN;
#pragma unroll
    for (int r = 0; r < 8; ++r) {
      int mr = r + 8 * h2;
      Ssm[mr][j0 + ml] = c[r] * 0.125f + bp[(size_t)mr * NN + j0 + ml];
    }
  }
  __syncthreads();
  if (t < 16) {                        // per-row max & sum
    float mx = -1e30f;
    for (int j = 0; j < NN; ++j) mx = fmaxf(mx, Ssm[t][j]);
    float s = 0.f;
    for (int j = 0; j < NN; ++j) s += expf(Ssm[t][j] - mx);
    rmax[t] = mx; rsum[t] = 1.f / s;
  }
  __syncthreads();
  for (int idx = t; idx < 16 * NN; idx += 256) {
    int r = idx >> 7, j = idx & 127;
    Pm[r][j] = (_Float16)(expf(Ssm[r][j] - rmax[r]) * rsum[r]);
  }
  __syncthreads();
  if (wave < 4) {                      // O tiles: wave w owns d tile d0 = 16w
    int d0 = wave * 16;
    v8f c = {};
#pragma unroll
    for (int j0 = 0; j0 < NN; j0 += 32) {
      v8h a0 = *(const v8h*)(&Pm[ml][j0 + 8 * h2]);
      v8h a1 = *(const v8h*)(&Pm[ml][j0 + 16 + 8 * h2]);
      v8h b0 = *(const v8h*)(&Vt[d0 + ml][j0 + 8 * h2]);
      v8h b1 = *(const v8h*)(&Vt[d0 + ml][j0 + 16 + 8 * h2]);
      c = __builtin_amdgcn_wmma_f32_16x16x32_f16(false, mk16(a0, a1), false, mk16(b0, b1),
                                                 (short)0, c, false, false);
    }
#pragma unroll
    for (int r = 0; r < 8; ++r)
      o[(rowb + it * 16 + r + 8 * h2) * INNER + hc + d0 + ml] = (_Float16)c[r];
  }
}

// ---------------- final head: mean over N, LN, @Wf + bf ----------------
__global__ void head_kernel(const float* __restrict__ x, const float* __restrict__ g,
                            const float* __restrict__ be, const float* __restrict__ Wf,
                            const float* __restrict__ bf, float* __restrict__ out) {
  __shared__ float red[128];
  int b = blockIdx.x, t = threadIdx.x;
  float p = 0.f;
  if (t < DLAT) {
    const float* xp = x + (size_t)b * NN * DLAT + t;
    for (int n = 0; n < NN; ++n) p += xp[(size_t)n * DLAT];
    p *= (1.f / NN);
  }
  red[t] = (t < DLAT) ? p : 0.f; __syncthreads();
  for (int s = 64; s > 0; s >>= 1) { if (t < s) red[t] += red[t + s]; __syncthreads(); }
  float mean = red[0] * (1.f / DLAT); __syncthreads();
  float d = (t < DLAT) ? p - mean : 0.f;
  red[t] = d * d; __syncthreads();
  for (int s = 64; s > 0; s >>= 1) { if (t < s) red[t] += red[t + s]; __syncthreads(); }
  float rstd = rsqrtf(red[0] * (1.f / DLAT) + 1e-5f); __syncthreads();
  float ln = (t < DLAT) ? (d * rstd * g[t] + be[t]) : 0.f;
  red[t] = (t < DLAT) ? ln * Wf[t] : 0.f; __syncthreads();
  for (int s = 64; s > 0; s >>= 1) { if (t < s) red[t] += red[t + s]; __syncthreads(); }
  if (t == 0) out[b] = red[0] + bf[0];
}

extern "C" void kernel_launch(void* const* d_in, const int* in_sizes, int n_in,
                              void* d_out, int out_size, void* d_ws, size_t ws_size,
                              hipStream_t stream) {
  const int*   sp_pos   = (const int*)d_in[0];
  const int*   edge_in  = (const int*)d_in[1];
  const int*   x_nodes  = (const int*)d_in[2];
  const int*   indeg    = (const int*)d_in[3];
  const int*   outdeg   = (const int*)d_in[4];
  const float* atom_emb = (const float*)d_in[5];
  const float* ideg_emb = (const float*)d_in[6];
  const float* odeg_emb = (const float*)d_in[7];
  const float* edge_emb = (const float*)d_in[8];
  const float* edge_dw  = (const float*)d_in[9];
  const float* sp_emb   = (const float*)d_in[10];
  const float* ln1_g    = (const float*)d_in[11];
  const float* ln1_b    = (const float*)d_in[12];
  const float* Wq       = (const float*)d_in[13];
  const float* Wkv      = (const float*)d_in[14];
  const float* Wo       = (const float*)d_in[15];
  const float* bo       = (const float*)d_in[16];
  const float* ln2_g    = (const float*)d_in[17];
  const float* ln2_b    = (const float*)d_in[18];
  const float* W1       = (const float*)d_in[19];
  const float* b1       = (const float*)d_in[20];
  const float* W2       = (const float*)d_in[21];
  const float* b2       = (const float*)d_in[22];
  const float* lnf_g    = (const float*)d_in[23];
  const float* lnf_b    = (const float*)d_in[24];
  const float* Wf       = (const float*)d_in[25];
  const float* bf       = (const float*)d_in[26];
  float* out = (float*)d_out;

  // ---- carve workspace (~55 MB) ----
  size_t off = 0;
  char* base = (char*)d_ws;
  auto take = [&](size_t bytes) -> char* {
    char* p = base + off; off += (bytes + 255) & ~(size_t)255; return p;
  };
  float*     biasb = (float*)take(sizeof(float) * BB * NN * NN);
  float*     xb    = (float*)take(sizeof(float) * MM * DLAT);
  _Float16*  hpad  = (_Float16*)take(sizeof(_Float16) * MM * KPAD);
  _Float16*  qbuf  = (_Float16*)take(sizeof(_Float16) * MM * INNER);
  _Float16*  kvbuf = (_Float16*)take(sizeof(_Float16) * MM * 2 * INNER);
  _Float16*  obuf  = (_Float16*)take(sizeof(_Float16) * MM * INNER);
  _Float16*  ffbuf = (_Float16*)take(sizeof(_Float16) * MM * FFD);
  _Float16*  WqT   = (_Float16*)take(sizeof(_Float16) * DEPTH * INNER * KPAD);
  _Float16*  WkvT  = (_Float16*)take(sizeof(_Float16) * DEPTH * 2 * INNER * KPAD);
  _Float16*  WoT   = (_Float16*)take(sizeof(_Float16) * DEPTH * DLAT * INNER);
  _Float16*  W1T   = (_Float16*)take(sizeof(_Float16) * DEPTH * FFD * KPAD);
  _Float16*  W2T   = (_Float16*)take(sizeof(_Float16) * DEPTH * DLAT * FFD);
  (void)ws_size; (void)n_in; (void)in_sizes; (void)out_size;

  // ---- one-time per call: weight convert, bias, embeddings ----
  {
    int tq = DEPTH * INNER * KPAD;
    wconv_kernel<<<(tq + 255) / 256, 256, 0, stream>>>(Wq, WqT, DLAT, KPAD, INNER, tq);
    int tkv = DEPTH * 2 * INNER * KPAD;
    wconv_kernel<<<(tkv + 255) / 256, 256, 0, stream>>>(Wkv, WkvT, DLAT, KPAD, 2 * INNER, tkv);
    int to = DEPTH * DLAT * INNER;
    wconv_kernel<<<(to + 255) / 256, 256, 0, stream>>>(Wo, WoT, INNER, INNER, DLAT, to);
    int t1 = DEPTH * FFD * KPAD;
    wconv_kernel<<<(t1 + 255) / 256, 256, 0, stream>>>(W1, W1T, DLAT, KPAD, FFD, t1);
    int t2 = DEPTH * DLAT * FFD;
    wconv_kernel<<<(t2 + 255) / 256, 256, 0, stream>>>(W2, W2T, FFD, FFD, DLAT, t2);
  }
  bias_kernel<<<(BB * NN * NN) / 256, 256, 0, stream>>>(sp_pos, edge_in, edge_emb, edge_dw,
                                                        sp_emb, biasb);
  embed_kernel<<<(MM * DLAT + 255) / 256, 256, 0, stream>>>(x_nodes, indeg, outdeg, atom_emb,
                                                            ideg_emb, odeg_emb, xb);

  // ---- 24 transformer layers ----
  for (int l = 0; l < DEPTH; ++l) {
    ln_rows_f16<<<MM, 128, 0, stream>>>(xb, ln1_g + l * DLAT, ln1_b + l * DLAT, hpad);
    gemm_f16_wmma<<<dim3(MM / 64, INNER / 64), 128, 0, stream>>>(
        hpad, KPAD, WqT + (size_t)l * INNER * KPAD, KPAD, INNER,
        nullptr, nullptr, qbuf, INNER, 0);
    gemm_f16_wmma<<<dim3(MM / 64, (2 * INNER) / 64), 128, 0, stream>>>(
        hpad, KPAD, WkvT + (size_t)l * 2 * INNER * KPAD, KPAD, 2 * INNER,
        nullptr, nullptr, kvbuf, 2 * INNER, 0);
    attn_kernel<<<BB * HEADS * (NN / 16), 256, 0, stream>>>(qbuf, kvbuf, biasb, obuf);
    gemm_f16_wmma<<<dim3(MM / 64, 2), 128, 0, stream>>>(
        obuf, INNER, WoT + (size_t)l * DLAT * INNER, INNER, DLAT,
        bo + l * DLAT, xb, nullptr, DLAT, 0);
    ln_rows_f16<<<MM, 128, 0, stream>>>(xb, ln2_g + l * DLAT, ln2_b + l * DLAT, hpad);
    gemm_f16_wmma<<<dim3(MM / 64, 3), 128, 0, stream>>>(
        hpad, KPAD, W1T + (size_t)l * FFD * KPAD, KPAD, FFD,
        b1 + l * FFD, nullptr, ffbuf, FFD, 1);
    gemm_f16_wmma<<<dim3(MM / 64, 2), 128, 0, stream>>>(
        ffbuf, FFD, W2T + (size_t)l * DLAT * FFD, FFD, DLAT,
        b2 + l * DLAT, xb, nullptr, DLAT, 0);
  }

  // ---- head ----
  head_kernel<<<BB, 128, 0, stream>>>(xb, lnf_g, lnf_b, Wf, bf, out);
}
